// ParticleNet_28621662060651
// MI455X (gfx1250) — matrix operand order repaired and checked by reading, not compiled, not run on hardware
//
#include <hip/hip_runtime.h>
#include <hip/hip_bf16.h>

// ParticleNet forward for MI455X (gfx1250).
// Compute-bound (~152 GFLOP of GEMM vs ~60us of HBM traffic at 23.3 TB/s),
// so all EdgeConv Linears run through v_wmma_f32_16x16x32_bf16 (bf16 in,
// f32 accumulate). GEMM: 128x64 block tile, 8 waves, 4 WMMA tiles per wave
// per 32-deep K-step. Workspace requirement: ~415 MB.

typedef unsigned short u16;
typedef unsigned int   u32;

typedef __attribute__((ext_vector_type(16))) __bf16 bf16x16;
typedef __attribute__((ext_vector_type(8)))  float  f32x8;

__device__ __forceinline__ u16 f2bf(float f) {
  u32 u = __float_as_uint(f);
  u32 r = (u + 0x7FFFu + ((u >> 16) & 1u)) >> 16;   // round-to-nearest-even
  return (u16)r;
}
__device__ __forceinline__ float bf2f(u16 h) {
  return __uint_as_float(((u32)h) << 16);
}

// ---------------------------------------------------------------- utilities
__global__ void pnet_copy_f(const float* __restrict__ src, float* __restrict__ dst, int total) {
  for (int i = blockIdx.x * blockDim.x + threadIdx.x; i < total; i += gridDim.x * blockDim.x)
    dst[i] = src[i];
}

// ---------------------------------------------------------------- kNN (K=16)
// One block per batch item, one thread per node. d2 row lives in LDS.
__global__ __launch_bounds__(128) void pnet_knn(const float* __restrict__ X,
                                                int Fstride, int Fknn,
                                                int* __restrict__ idx) {
  __shared__ float dist[128 * 128];                  // 64 KB (<= 320 KB/WGP)
  const int b = blockIdx.x;
  const int n = threadIdx.x;
  const float* xb = X + (size_t)b * 128 * Fstride;
  for (int m = 0; m < 128; ++m) {
    float acc = 0.f;
    for (int f = 0; f < Fknn; ++f) {
      float d = xb[n * Fstride + f] - xb[m * Fstride + f];
      acc += d * d;
    }
    dist[n * 128 + m] = (m == n) ? 3.0e38f : acc;    // exclude self-loop
  }
  // 16 smallest (set only; downstream mean over K is order-invariant)
  int* out = idx + ((size_t)b * 128 + n) * 16;
  for (int k = 0; k < 16; ++k) {
    float best = 3.0e38f;
    int bi = 0;
    for (int m = 0; m < 128; ++m) {
      float v = dist[n * 128 + m];
      if (v < best) { best = v; bi = m; }
    }
    dist[n * 128 + bi] = 3.0e38f;
    out[k] = bi;
  }
}

// ------------------------------------------------- EdgeConv message building
// M[e, f] = concat([x_i, x_j - x_i]) padded with zeros to Kpad, bf16.
__global__ void pnet_build_msg(const float* __restrict__ X, const int* __restrict__ idx,
                               u16* __restrict__ M, int Fcur, int Kpad, int total) {
  for (int i = blockIdx.x * blockDim.x + threadIdx.x; i < total; i += gridDim.x * blockDim.x) {
    int e = i / Kpad;
    int f = i - e * Kpad;
    int node = e >> 4;                 // b*128 + n
    float v = 0.f;
    if (f < Fcur) {
      v = X[(size_t)node * Fcur + f];
    } else if (f < 2 * Fcur) {
      int ff = f - Fcur;
      int m = idx[e];                  // neighbor index within batch
      int jn = (node & ~127) | m;      // b*128 + m
      v = X[(size_t)jn * Fcur + ff] - X[(size_t)node * Fcur + ff];
    }
    M[i] = f2bf(v);
  }
}

// ------------------------------------- weight conversion (transpose + pad)
// Wt[n, k] = bf16(W[k, n]), k padded to Kpad with zeros. Row stride = Kpad.
__global__ void pnet_convert_wt(const float* __restrict__ W, u16* __restrict__ Wt,
                                int Fi, int Kpad, int N) {
  int total = N * Kpad;
  for (int i = blockIdx.x * blockDim.x + threadIdx.x; i < total; i += gridDim.x * blockDim.x) {
    int n = i / Kpad;
    int k = i - n * Kpad;
    Wt[i] = (k < Fi) ? f2bf(W[(size_t)k * N + n]) : (u16)0;
  }
}

// --------------------------------------------------------------- bf16 GEMM
// H[E,N] = bf16(A[E,Kpad] x W[Kpad,N] + bias), W given K-major (Wt[N,Kpad]).
// Block = 8 waves computing a 128x64 tile; each wave owns 16(M) x 64(N):
// one A fragment is reused across 4 WMMAs per 32-deep K-step.
__global__ __launch_bounds__(256) void pnet_gemm_bf16(const u16* __restrict__ A,
                                                      const u16* __restrict__ Wt,
                                                      const float* __restrict__ bias,
                                                      u16* __restrict__ H,
                                                      int Kpad, int N) {
  // Padded row strides (20 dwords) keep fragment ds_load_b128s conflict-free.
  __shared__ u32 ldsA[128 * 20];       // 128 M-rows x 32 bf16 (dwords 0..15)
  __shared__ u32 ldsB[64 * 20];        // 64 N-rows  x 32 bf16 (K-contiguous)

  const int tid  = threadIdx.x;
  const int lane = tid & 31;
  const int wave = tid >> 5;           // 0..7
  const int g    = lane >> 4;          // half-wave select
  const int ln   = lane & 15;
  const int rowBase = blockIdx.x * 128;
  const int colBase = blockIdx.y * 64;

  // A staging: 2 threads per row, 32B (uint4 x2) each
  const int arow  = tid >> 1;          // 0..127
  const int ahalf = tid & 1;
  const u16* gA = A + (size_t)(rowBase + arow) * Kpad + ahalf * 16;
  uint4* lA = (uint4*)&ldsA[arow * 20 + ahalf * 8];

  // B staging: 4 threads per row, 16B (uint4) each
  const int brow = tid >> 2;           // 0..63
  const int bq   = tid & 3;
  const u16* gB = Wt + (size_t)(colBase + brow) * Kpad + bq * 8;
  uint4* lB = (uint4*)&ldsB[brow * 20 + bq * 4];

  union Frag { uint4 q[2]; bf16x16 v; };
  f32x8 acc[4];
#pragma unroll
  for (int j = 0; j < 4; ++j)
    acc[j] = (f32x8){0.f, 0.f, 0.f, 0.f, 0.f, 0.f, 0.f, 0.f};

  for (int k0 = 0; k0 < Kpad; k0 += 32) {
    // stage A tile (128x32 bf16) and B tile (64x32 bf16), both K-contiguous
    const uint4* ga4 = (const uint4*)gA;
    lA[0] = ga4[0];
    lA[1] = ga4[1];
    lB[0] = *(const uint4*)gB;
    if (k0 + 32 < Kpad) {
      __builtin_prefetch((const char*)(gA + 32), 0, 3);
      __builtin_prefetch((const char*)(gB + 32), 0, 3);
    }
    gA += 32;
    gB += 32;
    __syncthreads();

    Frag fa;
    const uint4* pa = (const uint4*)&ldsA[(wave * 16 + ln) * 20 + 4 * g];
    fa.q[0] = pa[0];                   // K = {0..7}  + 8g
    fa.q[1] = pa[2];                   // K = {16..23}+ 8g
#pragma unroll
    for (int j = 0; j < 4; ++j) {
      Frag fb;
      const uint4* pb = (const uint4*)&ldsB[(j * 16 + ln) * 20 + 4 * g];
      fb.q[0] = pb[0];                 // K = {0..7}  + 8g
      fb.q[1] = pb[2];                 // K = {16..23}+ 8g
      acc[j] = __builtin_amdgcn_wmma_f32_16x16x32_bf16(false, fa.v, false, fb.v,
                                                       (short)0, acc[j], false, false);
    }
    __syncthreads();
  }

  // C/D layout: VGPR r -> (M = r + 8g, N = ln)
#pragma unroll
  for (int j = 0; j < 4; ++j) {
    const int col = colBase + j * 16 + ln;
    const float bv = bias[col];
#pragma unroll
    for (int r = 0; r < 8; ++r) {
      int row = rowBase + wave * 16 + r + 8 * g;
      H[(size_t)row * N + col] = f2bf(acc[j][r] + bv);
    }
  }
}

// --------------------------------------- deterministic BatchNorm statistics
__global__ void pnet_reduce_partial(const u16* __restrict__ H, float* __restrict__ PART,
                                    int N, int rowsPerBlock) {
  int ch = threadIdx.x;
  if (ch >= N) return;
  int row0 = blockIdx.x * rowsPerBlock;
  float s = 0.f, s2 = 0.f;
  for (int r = 0; r < rowsPerBlock; ++r) {
    float v = bf2f(H[(size_t)(row0 + r) * N + ch]);
    s += v;
    s2 += v * v;
  }
  PART[(size_t)blockIdx.x * (2 * N) + ch] = s;
  PART[(size_t)blockIdx.x * (2 * N) + N + ch] = s2;
}

__global__ void pnet_stats_final(const float* __restrict__ PART, float* __restrict__ STATS,
                                 int twoN, int nblk) {
  int ch = threadIdx.x;
  if (ch >= twoN) return;
  float s = 0.f;
  for (int b = 0; b < nblk; ++b) s += PART[(size_t)b * twoN + ch];   // fixed order
  STATS[ch] = s;
}

__global__ void pnet_bn_relu(u16* __restrict__ H, const float* __restrict__ STATS,
                             const float* __restrict__ gamma, const float* __restrict__ beta,
                             int N, int total, float invE) {
  for (int i = blockIdx.x * blockDim.x + threadIdx.x; i < total; i += gridDim.x * blockDim.x) {
    int ch = i % N;
    float mean = STATS[ch] * invE;
    float var  = STATS[N + ch] * invE - mean * mean;
    float v = bf2f(H[i]);
    v = gamma[ch] * (v - mean) * rsqrtf(var + 1e-5f) + beta[ch];
    H[i] = f2bf(fmaxf(v, 0.f));
  }
}

// ------------------------------------------- mean over K + skip-concatenate
__global__ void pnet_aggregate(const u16* __restrict__ Hlast, const float* __restrict__ Xold,
                               float* __restrict__ Xnew, int Fo, int Fcur, int total) {
  int Fnew = Fo + Fcur;
  for (int i = blockIdx.x * blockDim.x + threadIdx.x; i < total; i += gridDim.x * blockDim.x) {
    int node = i / Fnew;
    int f = i - node * Fnew;
    float v;
    if (f < Fo) {
      float s = 0.f;
      for (int k = 0; k < 16; ++k)
        s += bf2f(Hlast[(size_t)(node * 16 + k) * Fo + f]);
      v = s * (1.f / 16.f);
    } else {
      v = Xold[(size_t)node * Fcur + (f - Fo)];
    }
    Xnew[i] = v;
  }
}

// ---------------------------------------------------------------- pool + FC
__global__ void pnet_pool(const float* __restrict__ X, float* __restrict__ P, int F, int total) {
  for (int i = blockIdx.x * blockDim.x + threadIdx.x; i < total; i += gridDim.x * blockDim.x) {
    int b = i / F;
    int f = i - b * F;
    float s = 0.f;
    for (int n = 0; n < 128; ++n) s += X[((size_t)b * 128 + n) * F + f];
    P[i] = s * (1.f / 128.f);
  }
}

__global__ __launch_bounds__(256) void pnet_fc1(const float* __restrict__ P,
                                                const float* __restrict__ W,
                                                const float* __restrict__ b,
                                                float* __restrict__ Hd) {
  int bb = blockIdx.x;
  int j = threadIdx.x;
  float s = b[j];
  for (int f = 0; f < 451; ++f) s += P[bb * 451 + f] * W[f * 256 + j];
  Hd[bb * 256 + j] = s;
}

__global__ void pnet_fc2(const float* __restrict__ Hd, const float* __restrict__ W,
                         const float* __restrict__ b, float* __restrict__ out) {
  int i = blockIdx.x * blockDim.x + threadIdx.x;
  if (i >= 128 * 5) return;
  int bb = i / 5;
  int o = i - bb * 5;
  float s = b[o];
  for (int j = 0; j < 256; ++j) s += Hd[bb * 256 + j] * W[j * 5 + o];
  out[i] = s;
}

// ---------------------------------------------------------------- dispatch
extern "C" void kernel_launch(void* const* d_in, const int* in_sizes, int n_in,
                              void* d_out, int out_size, void* d_ws, size_t ws_size,
                              hipStream_t stream) {
  (void)in_sizes; (void)n_in; (void)out_size; (void)ws_size;
  const int Bn = 128, Nn = 128;
  const int E = Bn * Nn * 16;                 // 262144 edges
  const int NODES = Bn * Nn;                  // 16384

  const float* x = (const float*)d_in[0];
  const float* fcW1 = (const float*)d_in[37];
  const float* fcb1 = (const float*)d_in[38];
  const float* fcW2 = (const float*)d_in[39];
  const float* fcb2 = (const float*)d_in[40];
  float* out = (float*)d_out;

  // workspace carve-out (requires ~415 MB)
  char* w = (char*)d_ws;
  size_t off = 0;
  auto take = [&](size_t bytes) -> void* {
    void* p = w + off;
    off += (bytes + 255) & ~(size_t)255;
    return p;
  };
  float* XA   = (float*)take((size_t)NODES * 451 * 4);
  float* XB   = (float*)take((size_t)NODES * 451 * 4);
  int*   IDX  = (int*)  take((size_t)E * 4);
  u16*   MSG  = (u16*)  take((size_t)E * 416 * 2);     // widest message matrix
  u16*   HB1  = (u16*)  take((size_t)E * 256 * 2);
  u16*   WBF  = (u16*)  take((size_t)416 * 256 * 2);
  float* PART = (float*)take((size_t)512 * 512 * 4);
  float* STATS= (float*)take(2048);
  float* POOL = (float*)take((size_t)Bn * 451 * 4);
  float* HID  = (float*)take((size_t)Bn * 256 * 4);
  u16*   HB2  = MSG;                                    // reuse MSG after layer 0

  pnet_copy_f<<<64, 256, 0, stream>>>(x, XA, NODES * 3);

  const int outs[3]  = {64, 128, 256};
  const int fins[3]  = {6, 134, 390};
  const int kpad0[3] = {32, 160, 416};

  float* Xcur = XA;
  float* Xnext = XB;
  int Fcur = 3;

  for (int cidx = 0; cidx < 3; ++cidx) {
    const int Fknn = (cidx == 0) ? 2 : Fcur;
    pnet_knn<<<Bn, 128, 0, stream>>>(Xcur, Fcur, Fknn, IDX);

    const int Kp0 = kpad0[cidx];
    pnet_build_msg<<<4096, 256, 0, stream>>>(Xcur, IDX, MSG, Fcur, Kp0, E * Kp0);

    const int N = outs[cidx];
    const u16* Ain[3]  = {MSG, HB1, HB2};
    u16*       Aout[3] = {HB1, HB2, HB1};
    const int  Kp[3]   = {Kp0, N, N};
    const int  Fi[3]   = {fins[cidx], N, N};

    for (int l = 0; l < 3; ++l) {
      const int base = 1 + (cidx * 3 + l) * 4;
      const float* W  = (const float*)d_in[base];
      const float* bb = (const float*)d_in[base + 1];
      const float* gg = (const float*)d_in[base + 2];
      const float* ee = (const float*)d_in[base + 3];

      pnet_convert_wt<<<256, 256, 0, stream>>>(W, WBF, Fi[l], Kp[l], N);

      dim3 grid(E / 128, N / 64);
      pnet_gemm_bf16<<<grid, 256, 0, stream>>>(Ain[l], WBF, bb, Aout[l], Kp[l], N);

      pnet_reduce_partial<<<512, 256, 0, stream>>>(Aout[l], PART, N, E / 512);
      pnet_stats_final<<<1, 512, 0, stream>>>(PART, STATS, 2 * N, 512);
      pnet_bn_relu<<<4096, 256, 0, stream>>>(Aout[l], STATS, gg, ee, N, E * N, 1.0f / E);
    }

    pnet_aggregate<<<2048, 256, 0, stream>>>(HB1, Xcur, Xnext, N, Fcur,
                                             NODES * (N + Fcur));
    Fcur += N;
    float* t = Xcur; Xcur = Xnext; Xnext = t;
  }

  pnet_pool<<<(Bn * 451 + 255) / 256, 256, 0, stream>>>(Xcur, POOL, 451, Bn * 451);
  pnet_fc1<<<Bn, 256, 0, stream>>>(POOL, fcW1, fcb1, HID);
  pnet_fc2<<<(Bn * 5 + 255) / 256, 256, 0, stream>>>(HID, fcW2, fcb2, out);
}
